// CoattentionModel_34943853920734
// MI455X (gfx1250) — compile-verified, matrix operand order checked by reading, not run on hardware
//
#include <hip/hip_runtime.h>

#define HW 3136
#define WIDE 56
#define CIN 512
#define KTOT 4608   // 512*9
#define NKT 144     // KTOT/32
#define CH 20
#define DLOW 10

typedef __attribute__((ext_vector_type(16))) __bf16 v16bf;
typedef __attribute__((ext_vector_type(8)))  __bf16 v8bf;
typedef __attribute__((ext_vector_type(8)))  float  v8f;

// ---------------- weight pack: fw[o][c][ky][kx] (f32) -> wBp[kt][n][kk] (bf16)
// k' = kt*32+kk enumerates K as r*512+c (tap-major), value = w[n][c][r]
__global__ __launch_bounds__(256) void k_pack_w(const float* __restrict__ w,
                                                __bf16* __restrict__ out) {
  int i = blockIdx.x * 256 + threadIdx.x;   // 144*512*32 = 2359296 exact
  int kk = i & 31;
  int n  = (i >> 5) & 511;
  int kt = i >> 14;
  int kp = kt * 32 + kk;
  int r  = kp >> 9;        // tap index ky*3+kx
  int c  = kp & 511;       // input channel
  out[i] = (__bf16)w[(n * 512 + c) * 9 + r];
}

__global__ __launch_bounds__(256) void k_cvt_bf16(const float* __restrict__ in,
                                                  __bf16* __restrict__ out, int n) {
  int i = blockIdx.x * 256 + threadIdx.x;
  if (i < n) out[i] = (__bf16)in[i];
}

// ---------------- 3x3 conv 512->512 pad1 + bias + ReLU, implicit-im2col bf16 WMMA GEMM
// act: [512][3136] bf16, wBp: [144][512][32] bf16 (fragment-ready), out: [512][3136] bf16
// Workgroup: 64M x 128N, 8 waves as (mi 0..1) x (ni 0..3), wave tile 32M x 32N (4 WMMA/step)
__global__ __launch_bounds__(256) void k_conv3x3_wmma(const __bf16* __restrict__ act,
                                                      const __bf16* __restrict__ wBp,
                                                      const float* __restrict__ bias,
                                                      __bf16* __restrict__ out) {
  __shared__ __bf16 As[64][40];   // 32 cols used; 80B row stride (16B aligned, bank-spread)
  const int tid  = threadIdx.x;
  const int lane = tid & 31;
  const int wv   = tid >> 5;
  const int mi   = wv & 1;
  const int ni   = wv >> 1;
  const int m0 = blockIdx.x * 64;
  const int n0 = blockIdx.y * 128;
  const int l15   = lane & 15;
  const int khalf = (lane >> 4) << 3;       // 0 or 8 (ISA 16-bit A/B lane-half split)

  // A-staging coords: thread -> (row mm, 8 contiguous k columns), all hoisted
  const int mm  = tid >> 2;                 // 0..63
  const int kkb = (tid & 3) * 8;            // 0,8,16,24
  const int m_st = m0 + mm;
  const int y_st = m_st / WIDE;
  const int x_st = m_st - y_st * WIDE;

  v8f acc00, acc01, acc10, acc11;           // [A-frag 0/1][B-frag 0/1]
  {
    float b0 = bias[n0 + ni * 32 + l15];
    float b1 = bias[n0 + ni * 32 + 16 + l15];
#pragma unroll
    for (int r = 0; r < 8; ++r) { acc00[r] = b0; acc10[r] = b0; acc01[r] = b1; acc11[r] = b1; }
  }

  const int arow0 = mi * 32 + l15;
  const int arow1 = arow0 + 16;
  const __bf16* wrow0 = wBp + (size_t)(n0 + ni * 32 + l15) * 32;
  const __bf16* wrow1 = wrow0 + 16 * 32;

  for (int kt = 0; kt < NKT; ++kt) {
    const int k0 = kt * 32;
    const int r  = k0 >> 9;                 // uniform tap 0..8
    const int c0 = k0 & 511;
    const int ky = (r * 11) >> 5;           // r/3 for r<9
    const int kx = r - ky * 3;

    // ---- stage A tile (branchless gather, one b128 LDS store per thread)
    {
      int iy = y_st + ky - 1, ix = x_st + kx - 1;
      bool valid = ((unsigned)iy < (unsigned)WIDE) & ((unsigned)ix < (unsigned)WIDE);
      int base = valid ? (iy * WIDE + ix) : 0;       // clamp to safe in-buffer addr
      const __bf16* src = act + (size_t)(c0 + kkb) * HW + base;
      v8bf tmp;
#pragma unroll
      for (int e = 0; e < 8; ++e) {
        __bf16 v = src[e * HW];                      // shared base + const ioffset
        tmp[e] = valid ? v : (__bf16)0.0f;
      }
      __syncthreads();                               // prior readers done
      *(v8bf*)&As[mm][kkb] = tmp;
      __syncthreads();                               // tile visible
    }

    // ---- fragments: A from LDS, B straight from fragment-ready global layout
    const __bf16* w0 = wrow0 + (size_t)kt * (512 * 32);
    const __bf16* w1 = wrow1 + (size_t)kt * (512 * 32);
    if (kt + 1 < NKT) __builtin_prefetch(w0 + 512 * 32, 0, 1);

    v8bf a0lo = *(const v8bf*)&As[arow0][khalf];
    v8bf a0hi = *(const v8bf*)&As[arow0][16 + khalf];
    v8bf a1lo = *(const v8bf*)&As[arow1][khalf];
    v8bf a1hi = *(const v8bf*)&As[arow1][16 + khalf];
    v8bf b0lo = *(const v8bf*)(w0 + khalf);
    v8bf b0hi = *(const v8bf*)(w0 + 16 + khalf);
    v8bf b1lo = *(const v8bf*)(w1 + khalf);
    v8bf b1hi = *(const v8bf*)(w1 + 16 + khalf);
    v16bf a0 = __builtin_shufflevector(a0lo, a0hi, 0,1,2,3,4,5,6,7,8,9,10,11,12,13,14,15);
    v16bf a1 = __builtin_shufflevector(a1lo, a1hi, 0,1,2,3,4,5,6,7,8,9,10,11,12,13,14,15);
    v16bf b0 = __builtin_shufflevector(b0lo, b0hi, 0,1,2,3,4,5,6,7,8,9,10,11,12,13,14,15);
    v16bf b1 = __builtin_shufflevector(b1lo, b1hi, 0,1,2,3,4,5,6,7,8,9,10,11,12,13,14,15);

    acc00 = __builtin_amdgcn_wmma_f32_16x16x32_bf16(false, a0, false, b0, (short)0, acc00, false, false);
    acc01 = __builtin_amdgcn_wmma_f32_16x16x32_bf16(false, a0, false, b1, (short)0, acc01, false, false);
    acc10 = __builtin_amdgcn_wmma_f32_16x16x32_bf16(false, a1, false, b0, (short)0, acc10, false, false);
    acc11 = __builtin_amdgcn_wmma_f32_16x16x32_bf16(false, a1, false, b1, (short)0, acc11, false, false);
  }

  // C/D layout: VGPR r -> M=r (lanes 0-15) / r+8 (lanes 16-31); lane%16 -> N
  const int mst = m0 + mi * 32 + ((lane >> 4) << 3);
  const int nc0 = n0 + ni * 32 + l15;
#pragma unroll
  for (int rr = 0; rr < 8; ++rr) {
    out[(size_t)nc0 * HW + mst + rr]             = (__bf16)fmaxf(acc00[rr], 0.0f);
    out[(size_t)(nc0 + 16) * HW + mst + rr]      = (__bf16)fmaxf(acc01[rr], 0.0f);
    out[(size_t)nc0 * HW + mst + 16 + rr]        = (__bf16)fmaxf(acc10[rr], 0.0f);
    out[(size_t)(nc0 + 16) * HW + mst + 16 + rr] = (__bf16)fmaxf(acc11[rr], 0.0f);
  }
}

// ---------------- 1x1 conv 512 -> 20 (no relu)
__global__ __launch_bounds__(256) void k_conv1x1(const __bf16* __restrict__ act,
                                                 const float* __restrict__ w,
                                                 const float* __restrict__ b,
                                                 float* __restrict__ out) {
  int t = blockIdx.x * 256 + threadIdx.x;   // 20*3136 exact
  int o = t / HW, p = t - o * HW;
  float s = b[o];
#pragma unroll 8
  for (int c = 0; c < CIN; ++c)
    s = fmaf((float)act[c * HW + p], w[o * CIN + c], s);
  out[t] = s;
}

// ---------------- global average pool (one channel per block)
__global__ __launch_bounds__(256) void k_gap(const float* __restrict__ x,
                                             float* __restrict__ out) {
  __shared__ float red[256];
  int c = blockIdx.x, tid = threadIdx.x;
  float s = 0.f;
  for (int p = tid; p < HW; p += 256) s += x[c * HW + p];
  red[tid] = s; __syncthreads();
  for (int o = 128; o > 0; o >>= 1) { if (tid < o) red[tid] += red[tid + o]; __syncthreads(); }
  if (tid == 0) out[c] = red[0] * (1.0f / HW);
}

// ---------------- low-rank projection
__global__ __launch_bounds__(256) void k_proj(const float* __restrict__ x,
                                              const float* __restrict__ w,
                                              const float* __restrict__ b,
                                              float* __restrict__ out) {
  int t = blockIdx.x * 256 + threadIdx.x;
  if (t >= DLOW * HW) return;
  int d = t / HW, p = t - d * HW;
  float s = b[d];
#pragma unroll
  for (int c = 0; c < CH; ++c) s = fmaf(x[c * HW + p], w[d * CH + c], s);
  out[t] = s;
}

// ---------------- co-attention (flash style): one wave per exemplar pixel
__global__ __launch_bounds__(256) void k_attention(const float* __restrict__ ef,
                                                   const float* __restrict__ qf,
                                                   const float* __restrict__ qfeat,
                                                   const float* __restrict__ gate_w,
                                                   float* __restrict__ out) {
  const int lane = threadIdx.x & 31;
  const int p = blockIdx.x * 8 + (threadIdx.x >> 5);   // 392*8 = 3136
  float e[DLOW];
#pragma unroll
  for (int d = 0; d < DLOW; ++d) e[d] = ef[d * HW + p];
  float m = -1e30f, Z = 0.f;
  float acc[CH];
#pragma unroll
  for (int c = 0; c < CH; ++c) acc[c] = 0.f;
  for (int q = lane; q < HW; q += 32) {
    float s = 0.f;
#pragma unroll
    for (int d = 0; d < DLOW; ++d) s = fmaf(e[d], qf[d * HW + q], s);
    float mn = fmaxf(m, s);
    float sc = __expf(m - mn);
    float wq = __expf(s - mn);
    Z = Z * sc + wq;
#pragma unroll
    for (int c = 0; c < CH; ++c) acc[c] = fmaf(wq, qfeat[c * HW + q], acc[c] * sc);
    m = mn;
  }
#pragma unroll
  for (int off = 16; off > 0; off >>= 1) {
    float m2 = __shfl_xor(m, off, 32);
    float Z2 = __shfl_xor(Z, off, 32);
    float mn = fmaxf(m, m2);
    float s1 = __expf(m - mn), s2 = __expf(m2 - mn);
    Z = Z * s1 + Z2 * s2;
#pragma unroll
    for (int c = 0; c < CH; ++c) acc[c] = acc[c] * s1 + __shfl_xor(acc[c], off, 32) * s2;
    m = mn;
  }
  float inv = 1.0f / Z;
  float g = 0.f;
  float att[CH];
#pragma unroll
  for (int c = 0; c < CH; ++c) { att[c] = acc[c] * inv; g = fmaf(gate_w[c], att[c], g); }
  float mask = 1.0f / (1.0f + __expf(-g));
  if (lane == 0) {
#pragma unroll
    for (int c = 0; c < CH; ++c) out[c * HW + p] = att[c] * mask;
  }
}

// ---------------- fusion conv 3x3, 40 -> 20, pad 1
__global__ __launch_bounds__(256) void k_fus_conv(const float* __restrict__ cat,
                                                  const float* __restrict__ w,
                                                  const float* __restrict__ b,
                                                  float* __restrict__ out) {
  int t = blockIdx.x * 256 + threadIdx.x;   // 20*3136 exact
  int o = t / HW, p = t - o * HW;
  int y = p / WIDE, x = p - y * WIDE;
  float s = b[o];
  for (int c = 0; c < 2 * CH; ++c) {
    const float* wc = w + (o * 2 * CH + c) * 9;
    const float* xc = cat + c * HW;
#pragma unroll
    for (int ky = 0; ky < 3; ++ky) {
      int iy = y + ky - 1;
      if ((unsigned)iy >= (unsigned)WIDE) continue;
#pragma unroll
      for (int kx = 0; kx < 3; ++kx) {
        int ix = x + kx - 1;
        if ((unsigned)ix >= (unsigned)WIDE) continue;
        s = fmaf(xc[iy * WIDE + ix], wc[ky * 3 + kx], s);
      }
    }
  }
  out[t] = s;
}

// ---------------- GRU gates z,r
__global__ __launch_bounds__(256) void k_gru_zr(const float* __restrict__ inp,
                                                const float* __restrict__ h,
                                                const float* __restrict__ rw, const float* __restrict__ rb,
                                                const float* __restrict__ zw, const float* __restrict__ zb,
                                                float* __restrict__ zbuf, float* __restrict__ rbuf) {
  int t = blockIdx.x * 256 + threadIdx.x;   // 20*3136 exact
  int o = t / HW, p = t - o * HW;
  float sz = zb[o], sr = rb[o];
  for (int c = 0; c < CH; ++c) {
    float xv = inp[c * HW + p], hv = h[c * HW + p];
    sz = fmaf(zw[o * 2 * CH + c], xv, sz);
    sr = fmaf(rw[o * 2 * CH + c], xv, sr);
    sz = fmaf(zw[o * 2 * CH + CH + c], hv, sz);
    sr = fmaf(rw[o * 2 * CH + CH + c], hv, sr);
  }
  zbuf[t] = 1.0f / (1.0f + __expf(-sz));
  rbuf[t] = 1.0f / (1.0f + __expf(-sr));
}

// ---------------- GRU output
__global__ __launch_bounds__(256) void k_gru_out(const float* __restrict__ inp,
                                                 const float* __restrict__ h,
                                                 const float* __restrict__ ow, const float* __restrict__ ob,
                                                 const float* __restrict__ zbuf, const float* __restrict__ rbuf,
                                                 float* __restrict__ out) {
  int t = blockIdx.x * 256 + threadIdx.x;   // 20*3136 exact
  int o = t / HW, p = t - o * HW;
  float so = ob[o];
  for (int c = 0; c < CH; ++c) {
    so = fmaf(ow[o * 2 * CH + c], inp[c * HW + p], so);
    so = fmaf(ow[o * 2 * CH + CH + c], rbuf[c * HW + p] * h[c * HW + p], so);
  }
  float z = zbuf[t];
  out[t] = h[o * HW + p] * (1.0f - z) + tanhf(so) * z;
}

// ---------------- group dropout layer (single block: global max over att map)
__global__ __launch_bounds__(256) void k_gdl(const float* __restrict__ x,
                                             const float* __restrict__ rnd, int ridx,
                                             float* __restrict__ out) {
  __shared__ float att[HW];
  __shared__ float red[256];
  int tid = threadIdx.x;
  float lmax = -1e30f;
  for (int p = tid; p < HW; p += 256) {
    float s = 0.f;
#pragma unroll
    for (int c = 0; c < CH; ++c) s += x[c * HW + p];
    s *= (1.0f / CH);
    att[p] = s;
    lmax = fmaxf(lmax, s);
  }
  red[tid] = lmax; __syncthreads();
  for (int o = 128; o > 0; o >>= 1) { if (tid < o) red[tid] = fmaxf(red[tid], red[tid + o]); __syncthreads(); }
  float mx = red[0];
  float binary = floorf(rnd[ridx] + 0.8f);
  float th = mx * 0.7f;
  for (int i = tid; i < CH * HW; i += 256) {
    int p = i % HW;
    float a = att[p];
    float imp = 1.0f / (1.0f + __expf(-a));
    float dm = (a < th) ? 1.0f : 0.0f;
    float sel = (1.0f - binary) * imp + binary * dm;
    float v = x[i];
    out[i] = (v * sel + v) * 0.5f;
  }
}

extern "C" void kernel_launch(void* const* d_in, const int* in_sizes, int n_in,
                              void* d_out, int out_size, void* d_ws, size_t ws_size,
                              hipStream_t stream) {
  (void)in_sizes; (void)n_in; (void)out_size; (void)ws_size;
  const float* input[3] = {(const float*)d_in[0], (const float*)d_in[1], (const float*)d_in[2]};
  const float* gdl_rand = (const float*)d_in[3];
  const float* fw[3] = {(const float*)d_in[4], (const float*)d_in[6], (const float*)d_in[8]};
  const float* fb[3] = {(const float*)d_in[5], (const float*)d_in[7], (const float*)d_in[9]};
  const float* fw4 = (const float*)d_in[10];
  const float* fb4 = (const float*)d_in[11];
  const float* proja_w = (const float*)d_in[12];
  const float* proja_b = (const float*)d_in[13];
  const float* projb_w = (const float*)d_in[14];
  const float* projb_b = (const float*)d_in[15];
  const float* gate_w  = (const float*)d_in[16];
  const float* fus_w   = (const float*)d_in[17];
  const float* fus_b   = (const float*)d_in[18];
  const float* gru_rw  = (const float*)d_in[19];
  const float* gru_rb  = (const float*)d_in[20];
  const float* gru_zw  = (const float*)d_in[21];
  const float* gru_zb  = (const float*)d_in[22];
  const float* gru_ow  = (const float*)d_in[23];
  const float* gru_ob  = (const float*)d_in[24];
  float* out = (float*)d_out;

  char* ws = (char*)d_ws;
  size_t off = 0;
  auto take = [&](size_t bytes) -> void* {
    void* p = ws + off;
    off += (bytes + 255) & ~(size_t)255;
    return p;
  };
  __bf16* bw[3];
  for (int i = 0; i < 3; ++i) bw[i] = (__bf16*)take((size_t)KTOT * 512 * 2);
  __bf16* actA = (__bf16*)take((size_t)CIN * HW * 2);
  __bf16* actB = (__bf16*)take((size_t)CIN * HW * 2);
  float* st[3]; for (int i = 0; i < 3; ++i) st[i] = (float*)take((size_t)CH * HW * 4);
  float* aa[3]; for (int i = 0; i < 3; ++i) aa[i] = (float*)take((size_t)CH * HW * 4);
  float* zbuf = (float*)take((size_t)CH * HW * 4);
  float* rbuf = (float*)take((size_t)CH * HW * 4);
  float* tmpG = (float*)take((size_t)CH * HW * 4);
  float* ef   = (float*)take((size_t)DLOW * HW * 4);
  float* qf   = (float*)take((size_t)DLOW * HW * 4);
  float* cat  = (float*)take((size_t)2 * CH * HW * 4);

  for (int i = 0; i < 3; ++i)
    k_pack_w<<<9216, 256, 0, stream>>>(fw[i], bw[i]);

  for (int img = 0; img < 3; ++img) {
    k_cvt_bf16<<<6272, 256, 0, stream>>>(input[img], actA, CIN * HW);
    k_conv3x3_wmma<<<dim3(49, 4), 256, 0, stream>>>(actA, bw[0], fb[0], actB);
    k_conv3x3_wmma<<<dim3(49, 4), 256, 0, stream>>>(actB, bw[1], fb[1], actA);
    k_conv3x3_wmma<<<dim3(49, 4), 256, 0, stream>>>(actA, bw[2], fb[2], actB);
    k_conv1x1<<<245, 256, 0, stream>>>(actB, fw4, fb4, st[img]);
    k_gap<<<20, 256, 0, stream>>>(st[img], out + 40 * img);   // x1ss/x2ss/x3ss
  }

  const int trip[3][3] = {{0, 1, 2}, {1, 0, 2}, {2, 0, 1}};
  for (int r = 0; r < 3; ++r) {
    for (int t = 0; t < 3; ++t) {
      int s0 = trip[t][0], s1 = trip[t][1], s2 = trip[t][2];
      k_proj<<<123, 256, 0, stream>>>(st[s0], proja_w, proja_b, ef);
      k_proj<<<123, 256, 0, stream>>>(st[s1], projb_w, projb_b, qf);
      k_attention<<<392, 256, 0, stream>>>(ef, qf, st[s1], gate_w, cat);
      k_proj<<<123, 256, 0, stream>>>(st[s0], proja_w, proja_b, ef);
      k_proj<<<123, 256, 0, stream>>>(st[s2], projb_w, projb_b, qf);
      k_attention<<<392, 256, 0, stream>>>(ef, qf, st[s2], gate_w, cat + CH * HW);
      k_fus_conv<<<245, 256, 0, stream>>>(cat, fus_w, fus_b, aa[t]);
    }
    for (int t = 0; t < 3; ++t) {
      k_gru_zr<<<245, 256, 0, stream>>>(aa[t], st[t], gru_rw, gru_rb, gru_zw, gru_zb, zbuf, rbuf);
      k_gru_out<<<245, 256, 0, stream>>>(aa[t], st[t], gru_ow, gru_ob, zbuf, rbuf, tmpG);
      k_gdl<<<1, 256, 0, stream>>>(tmpG, gdl_rand, 3 * r + t, st[t]);
    }
  }

  k_gap<<<20, 256, 0, stream>>>(st[0], out + 20);
  k_gap<<<20, 256, 0, stream>>>(st[1], out + 60);
  k_gap<<<20, 256, 0, stream>>>(st[2], out + 100);
}
